// multi_head_attention_13443247636922
// MI455X (gfx1250) — matrix operand order repaired and checked by reading, not compiled, not run on hardware
//
#include <hip/hip_runtime.h>

#define E_DIM 1024
#define H_DIM 16
#define W_DIM 1024
#define B_DIM 4
#define D_DIM 64
#define EE ((size_t)E_DIM * E_DIM)
#define EW ((size_t)E_DIM * W_DIM)

typedef __attribute__((ext_vector_type(16))) _Float16 v16h;
typedef __attribute__((ext_vector_type(8)))  _Float16 h8;
typedef __attribute__((ext_vector_type(8)))  float    v8f;

union V16U { v16h v; h8 h[2]; };

// A-operand (16x32 f16): lane = row (lane&15); per-lane halves:
//   elements 0..15 = K[ 2c .. 2c+15 ]   (c = lane<16 ? 0 : 8)
__device__ __forceinline__ v16h load_a16(const _Float16* base, int row, int ld, int k, int c) {
    const _Float16* p = base + (size_t)row * ld + k + 2 * c;
    V16U u;
    u.h[0] = *(const h8*)(p);
    u.h[1] = *(const h8*)(p + 8);
    return u.v;
}

// B-operand (32x16 f16): lane&15 = column; lanes 0-15 hold K=0..15,
// lanes 16-31 hold K=16..31 -> one contiguous 16-half chunk per lane.
__device__ __forceinline__ v16h load_b16(const _Float16* base, int row, int ld, int k) {
    return *(const v16h*)(base + (size_t)row * ld + k);
}

__device__ __forceinline__ v8f wmma_f16(v16h a, v16h b, v8f c) {
    return __builtin_amdgcn_wmma_f32_16x16x32_f16(false, a, false, b, (short)0, c, false, false);
}

// ---------------------------------------------------------------------------
// 1) f32 -> f16 weight conversion: Lh = [L_K, L_Q, L_V, M] (A-operand layout)
// ---------------------------------------------------------------------------
__global__ __launch_bounds__(256)
void cvt_weights_kernel(const float* __restrict__ lk, const float* __restrict__ lq,
                        const float* __restrict__ lv, const float* __restrict__ mw,
                        _Float16* __restrict__ Lh) {
    size_t i = (size_t)blockIdx.x * blockDim.x + threadIdx.x;
    size_t stride = (size_t)gridDim.x * blockDim.x;
    for (; i < EE; i += stride) {
        Lh[i]          = (_Float16)lk[i];
        Lh[EE + i]     = (_Float16)lq[i];
        Lh[2 * EE + i] = (_Float16)lv[i];
        Lh[3 * EE + i] = (_Float16)mw[i];
    }
}

// ---------------------------------------------------------------------------
// 2) convert + transpose x[z] (E x W f32) -> Xt[z] (W x E f16), e contiguous
// ---------------------------------------------------------------------------
__global__ __launch_bounds__(256)
void cvt_transpose_kernel(const float* __restrict__ x, _Float16* __restrict__ Xt) {
    __shared__ _Float16 tile[32][33];
    const int z = blockIdx.z;
    const float* in = x + (size_t)z * EW;
    _Float16* out = Xt + (size_t)z * EW;
    const int tx = threadIdx.x, ty = threadIdx.y;
    const int w0 = blockIdx.x * 32, e0 = blockIdx.y * 32;
#pragma unroll
    for (int j = 0; j < 32; j += 8)
        tile[ty + j][tx] = (_Float16)in[(size_t)(e0 + ty + j) * W_DIM + w0 + tx];
    __syncthreads();
#pragma unroll
    for (int j = 0; j < 32; j += 8)
        out[(size_t)(w0 + ty + j) * E_DIM + e0 + tx] = tile[tx][ty + j];
}

// ---------------------------------------------------------------------------
// 3) projection GEMM: C = L_t @ x[t,b]  (1024x1024x1024, f16 in, f32 acc)
//    Wave tile 64x32 (4 m-tiles x 2 n-tiles): 8 WMMA per 12 b128 loads.
//    t=0 -> Kt[b,h,w,d], t=1 -> Qt[b,h,w,d] (head-transposed), t=2 -> Vv[b,e,w]
// ---------------------------------------------------------------------------
__global__ __launch_bounds__(256)
void proj_gemm_kernel(const _Float16* __restrict__ Lh, const _Float16* __restrict__ Xt,
                      _Float16* __restrict__ Kt, _Float16* __restrict__ Qt,
                      _Float16* __restrict__ Vv) {
    const int z = blockIdx.y;           // z = t*B + b
    const int t = z >> 2;
    const int b = z & 3;
    const int wid  = threadIdx.x >> 5;
    const int lane = threadIdx.x & 31;
    const int l15  = lane & 15;
    const int c    = (lane < 16) ? 0 : 8;
    const int bhi  = (lane < 16) ? 0 : 16;
    const int roff = (lane < 16) ? 0 : 8;
    const int gw = blockIdx.x * 8 + wid;            // 0..511 wave tiles
    const int m0 = (gw >> 5) * 64, n0 = (gw & 31) * 32;

    const _Float16* A  = Lh + (size_t)t * EE;
    const _Float16* Bt = Xt + (size_t)z * EW;       // [w][e] (K contiguous)

    v8f acc[4][2] = {};
    for (int k = 0; k < E_DIM; k += 32) {
        v16h bv[2], av[4];
#pragma unroll
        for (int ni = 0; ni < 2; ++ni)
            bv[ni] = load_b16(Bt, n0 + 16 * ni + l15, E_DIM, k + bhi);
#pragma unroll
        for (int mi = 0; mi < 4; ++mi)
            av[mi] = load_a16(A, m0 + 16 * mi + l15, E_DIM, k, c);
#pragma unroll
        for (int mi = 0; mi < 4; ++mi)
#pragma unroll
            for (int ni = 0; ni < 2; ++ni)
                acc[mi][ni] = wmma_f16(av[mi], bv[ni], acc[mi][ni]);
    }

    if (t < 2) {
        _Float16* Out = ((t == 0) ? Kt : Qt) + (size_t)b * EW;   // [h][w][d]
#pragma unroll
        for (int mi = 0; mi < 4; ++mi)
#pragma unroll
            for (int ni = 0; ni < 2; ++ni) {
                const int e0 = m0 + 16 * mi;
                const int h = e0 >> 6;
                const int dbase = (e0 & 63) + roff;
                const int w = n0 + 16 * ni + l15;
                h8 hv;
#pragma unroll
                for (int r = 0; r < 8; ++r) hv[r] = (_Float16)acc[mi][ni][r];
                *(h8*)(Out + ((size_t)h * W_DIM + w) * D_DIM + dbase) = hv;
            }
    } else {
        _Float16* Out = Vv + (size_t)b * EW;                     // [e][w]
#pragma unroll
        for (int mi = 0; mi < 4; ++mi)
#pragma unroll
            for (int ni = 0; ni < 2; ++ni) {
                const int e0 = m0 + 16 * mi + roff;
                const int w = n0 + 16 * ni + l15;
#pragma unroll
                for (int r = 0; r < 8; ++r)
                    Out[(size_t)(e0 + r) * W_DIM + w] = (_Float16)acc[mi][ni][r];
            }
    }
}

// ---------------------------------------------------------------------------
// 4) attention: per (b,h), flash-style online softmax over q (the reduction
//    axis of PV), causal mask q>k, column softmax, /sqrt(E).
//    1 wave = 16 k-columns; block = 8 waves = 128 columns.
//    att[b][w][e] f16 output (B-operand layout for the final GEMM).
// ---------------------------------------------------------------------------
__global__ __launch_bounds__(256)
void attn_kernel(const _Float16* __restrict__ Qt, const _Float16* __restrict__ Kt,
                 const _Float16* __restrict__ Vv, _Float16* __restrict__ att) {
    __shared__ _Float16 plds[8][16][32];      // per-wave p tile, [k][q] layout
    const int bh = blockIdx.y;                // b*H + h
    const int b = bh >> 4;
    const int h = bh & 15;
    const int wid  = threadIdx.x >> 5;
    const int lane = threadIdx.x & 31;
    const int l15  = lane & 15;
    const int c    = (lane < 16) ? 0 : 8;
    const int bhi  = (lane < 16) ? 0 : 16;
    const int roff = (lane < 16) ? 0 : 8;

    const int k0 = blockIdx.x * 128 + wid * 16;
    const int kcol = k0 + l15;

    const _Float16* Qbh = Qt + (size_t)bh * W_DIM * D_DIM;         // [w][d]
    const _Float16* Kbh = Kt + (size_t)bh * W_DIM * D_DIM;         // [w][d]
    const _Float16* Vbh = Vv + (size_t)bh * (size_t)D_DIM * W_DIM; // [d][w]

    v8f acc[4] = {};
    float m_run = -1e30f;
    float l_run = 0.0f;

    const int qend = k0 + 16;
    for (int q0 = 0; q0 < qend; q0 += 32) {
        // S[32q x 16k] = Q^T K  (inner d = 64 -> 2 WMMA steps)
        v8f s[2] = {};
#pragma unroll
        for (int dstep = 0; dstep < 64; dstep += 32) {
            v16h bk  = load_b16(Kbh, k0 + l15, D_DIM, dstep + bhi);
            v16h aq0 = load_a16(Qbh, q0 + l15,      D_DIM, dstep, c);
            v16h aq1 = load_a16(Qbh, q0 + 16 + l15, D_DIM, dstep, c);
            s[0] = wmma_f16(aq0, bk, s[0]);
            s[1] = wmma_f16(aq1, bk, s[1]);
        }
        // causal mask + block max (column = kcol; rows split across half-waves)
        float bm = -1e30f;
#pragma unroll
        for (int qt = 0; qt < 2; ++qt)
#pragma unroll
            for (int r = 0; r < 8; ++r) {
                const int q = q0 + 16 * qt + roff + r;
                const float sv = (q <= kcol) ? s[qt][r] : -1e30f;
                s[qt][r] = sv;
                bm = fmaxf(bm, sv);
            }
        bm = fmaxf(bm, __shfl_xor(bm, 16, 32));
        const float m_new = fmaxf(m_run, bm);
        const float scale = __expf(m_run - m_new);

        // p = exp(s - m_new); write to LDS in [k][q] (B-operand) layout
        float rs = 0.0f;
#pragma unroll
        for (int qt = 0; qt < 2; ++qt) {
            h8 ph;
#pragma unroll
            for (int r = 0; r < 8; ++r) {
                const float pv = __expf(s[qt][r] - m_new);
                rs += pv;
                ph[r] = (_Float16)pv;
            }
            *(h8*)(&plds[wid][l15][16 * qt + roff]) = ph;
        }
        rs += __shfl_xor(rs, 16, 32);
        l_run = l_run * scale + rs;
        m_run = m_new;

#pragma unroll
        for (int t = 0; t < 4; ++t)
#pragma unroll
            for (int r = 0; r < 8; ++r) acc[t][r] *= scale;

        asm volatile("s_wait_dscnt 0x0" ::: "memory");   // own-wave LDS WAR/RAW
        const v16h bp = *(const v16h*)(&plds[wid][l15][bhi]);
#pragma unroll
        for (int t = 0; t < 4; ++t) {
            v16h av = load_a16(Vbh, 16 * t + l15, W_DIM, q0, c);
            acc[t] = wmma_f16(av, bp, acc[t]);
        }
    }

    const float inv = 1.0f / (l_run * 32.0f);   // softmax normalize * E^-0.5
    _Float16* ob = att + ((size_t)b * W_DIM + kcol) * E_DIM + h * D_DIM;
#pragma unroll
    for (int t = 0; t < 4; ++t) {
        h8 hv;
#pragma unroll
        for (int r = 0; r < 8; ++r) hv[r] = (_Float16)(acc[t][r] * inv);
        *(h8*)(ob + 16 * t + roff) = hv;
    }
}

// ---------------------------------------------------------------------------
// 5) final GEMM: Y[b] = M @ out[b] + bias   (f32 output, 64x32 wave tile)
// ---------------------------------------------------------------------------
__global__ __launch_bounds__(256)
void out_gemm_kernel(const _Float16* __restrict__ Mh, const _Float16* __restrict__ att,
                     const float* __restrict__ bias, float* __restrict__ Y) {
    const int b = blockIdx.y;
    const int wid  = threadIdx.x >> 5;
    const int lane = threadIdx.x & 31;
    const int l15  = lane & 15;
    const int c    = (lane < 16) ? 0 : 8;
    const int bhi  = (lane < 16) ? 0 : 16;
    const int roff = (lane < 16) ? 0 : 8;
    const int gw = blockIdx.x * 8 + wid;            // 0..511
    const int m0 = (gw >> 5) * 64, n0 = (gw & 31) * 32;

    const _Float16* Bt = att + (size_t)b * EW;      // [w][e]
    v8f acc[4][2] = {};
    for (int k = 0; k < E_DIM; k += 32) {
        v16h bv[2], av[4];
#pragma unroll
        for (int ni = 0; ni < 2; ++ni)
            bv[ni] = load_b16(Bt, n0 + 16 * ni + l15, E_DIM, k + bhi);
#pragma unroll
        for (int mi = 0; mi < 4; ++mi)
            av[mi] = load_a16(Mh, m0 + 16 * mi + l15, E_DIM, k, c);
#pragma unroll
        for (int mi = 0; mi < 4; ++mi)
#pragma unroll
            for (int ni = 0; ni < 2; ++ni)
                acc[mi][ni] = wmma_f16(av[mi], bv[ni], acc[mi][ni]);
    }
    float* Yb = Y + (size_t)b * EW;
#pragma unroll
    for (int mi = 0; mi < 4; ++mi)
#pragma unroll
        for (int ni = 0; ni < 2; ++ni) {
            const int o0 = m0 + 16 * mi + roff;
            const int w = n0 + 16 * ni + l15;
#pragma unroll
            for (int r = 0; r < 8; ++r)
                Yb[(size_t)(o0 + r) * W_DIM + w] = acc[mi][ni][r] + bias[o0 + r];
        }
}

// ---------------------------------------------------------------------------
extern "C" void kernel_launch(void* const* d_in, const int* in_sizes, int n_in,
                              void* d_out, int out_size, void* d_ws, size_t ws_size,
                              hipStream_t stream) {
    const float* x    = (const float*)d_in[0];   // [3,B,E,W]
    const float* LQ   = (const float*)d_in[1];
    const float* LK   = (const float*)d_in[2];
    const float* LV   = (const float*)d_in[3];
    const float* Mw   = (const float*)d_in[4];
    const float* bias = (const float*)d_in[5];   // [E]
    float* Y = (float*)d_out;                    // [B,E,W]

    _Float16* Lh  = (_Float16*)d_ws;             // 4*EE   halfs  (L_K,L_Q,L_V,M)
    _Float16* Xt  = Lh + 4 * EE;                 // 12*EW  halfs  x transposed f16
    _Float16* Kt  = Xt + 12 * EW;                // 4*EW   [b,h,w,d]
    _Float16* Qt  = Kt + 4 * EW;                 // 4*EW   [b,h,w,d]
    _Float16* Vv  = Qt + 4 * EW;                 // 4*EW   [b,h,d,w]
    _Float16* att = Vv + 4 * EW;                 // 4*EW   [b,w,e]

    hipLaunchKernelGGL(cvt_weights_kernel, dim3(2048), dim3(256), 0, stream,
                       LK, LQ, LV, Mw, Lh);
    hipLaunchKernelGGL(cvt_transpose_kernel, dim3(32, 32, 12), dim3(32, 8), 0, stream,
                       x, Xt);
    hipLaunchKernelGGL(proj_gemm_kernel, dim3(64, 12), dim3(256), 0, stream,
                       Lh, Xt, Kt, Qt, Vv);
    hipLaunchKernelGGL(attn_kernel, dim3(8, 64), dim3(256), 0, stream,
                       Qt, Kt, Vv, att);
    hipLaunchKernelGGL(out_gemm_kernel, dim3(64, 4), dim3(256), 0, stream,
                       Lh + 3 * EE, att, bias, Y);
}
